// QMamba_17849884082755
// MI455X (gfx1250) — matrix operand order repaired and checked by reading, not compiled, not run on hardware
//
#include <hip/hip_runtime.h>
#include <hip/hip_bf16.h>

#define DD 128
#define CK 384          // 3 taps * 128 in-channels, c = tap*128 + i
#define SAMP (DD*DD)
#define NBT 800
#define EPSF 1e-12f
#define INV_SQRT2F 0.70710678118654752440f

typedef __attribute__((ext_vector_type(16))) __bf16 v16bf;
typedef __attribute__((ext_vector_type(8)))  float v8f;
typedef unsigned int v4u __attribute__((ext_vector_type(4)));

union BF16x16 { unsigned short u[16]; v4u q[2]; v16bf v; };

__device__ __forceinline__ unsigned short f2bf(float f) {
    unsigned int u = __float_as_uint(f);
    unsigned int r = (u + 0x7FFFu + ((u >> 16) & 1u)) >> 16;
    return (unsigned short)r;
}
__device__ __forceinline__ float bf2f(unsigned short h) {
    return __uint_as_float(((unsigned int)h) << 16);
}

// async global->LDS copy: each active lane moves 16 bytes (ASYNCcnt-tracked)
__device__ __forceinline__ void async_b128(void* lds_ptr, const void* gaddr) {
    unsigned lofs = (unsigned)(size_t)lds_ptr;   // low 32 bits of flat shared addr == LDS offset
    asm volatile("global_load_async_to_lds_b128 %0, %1, off"
                 :: "v"(lofs), "v"(gaddr) : "memory");
}
__device__ __forceinline__ void async_wait0() {
    asm volatile("s_wait_asynccnt 0" ::: "memory");
}

__device__ float blockReduceSum128(float v, float* sbuf) {
    int tid = threadIdx.x;
    sbuf[tid] = v;
    __syncthreads();
    for (int s = 64; s > 0; s >>= 1) {
        if (tid < s) sbuf[tid] += sbuf[tid + s];
        __syncthreads();
    }
    float r = sbuf[0];
    __syncthreads();
    return r;
}

// ---------------- lang vectors ----------------
__global__ __launch_bounds__(128) void lang_kernel(
    const float* __restrict__ l1, const float* __restrict__ l2,
    const float* __restrict__ Wl1, const float* __restrict__ bl1,
    const float* __restrict__ Wl2, const float* __restrict__ bl2,
    float* __restrict__ la) {
    __shared__ float sbuf[128];
    int d = threadIdx.x;
    float a = bl1[d];
    for (int k = 0; k < 768; ++k) a += l1[k] * Wl1[k * DD + d];
    a = fmaxf(a, 0.f);
    float n = sqrtf(blockReduceSum128(a * a, sbuf));
    la[d] = a / fmaxf(n, EPSF);
    float b = bl2[d];
    for (int k = 0; k < 768; ++k) b += l2[k] * Wl2[k * DD + d];
    b = fmaxf(b, 0.f);
    float n2 = sqrtf(blockReduceSum128(b * b, sbuf));
    la[DD + d] = b / fmaxf(n2, EPSF);
}

// ---------------- per-timestep prep ----------------
__global__ __launch_bounds__(128) void prep_kernel(
    const float* __restrict__ text, const float* __restrict__ audio,
    const float* __restrict__ visual, const float* __restrict__ beats,
    const float* __restrict__ Wt, const float* __restrict__ bt_,
    const float* __restrict__ Wa, const float* __restrict__ ba,
    const float* __restrict__ Wv, const float* __restrict__ bv,
    const float* __restrict__ Wb_, const float* __restrict__ bb,
    const float* __restrict__ la,
    float* __restrict__ entr, float* __restrict__ enti, float* __restrict__ wmod) {
    __shared__ float sbuf[128];
    int bt = blockIdx.x, d = threadIdx.x;
    const float* tx = text   + (size_t)bt * 768;
    const float* au = audio  + (size_t)bt * 74;
    const float* vz = visual + (size_t)bt * 35;
    const float* be = beats  + (size_t)bt * 768;

    float a0 = bt_[d], a3 = bb[d];
    for (int k = 0; k < 768; ++k) {
        float tk = tx[k], bk = be[k];
        a0 = fmaf(tk, Wt[k * DD + d], a0);
        a3 = fmaf(bk, Wb_[k * DD + d], a3);
    }
    float a1 = ba[d];
    for (int k = 0; k < 74; ++k) a1 = fmaf(au[k], Wa[k * DD + d], a1);
    float a2 = bv[d];
    for (int k = 0; k < 35; ++k) a2 = fmaf(vz[k], Wv[k * DD + d], a2);

    a0 = fmaxf(a0, 0.f); a1 = fmaxf(a1, 0.f); a2 = fmaxf(a2, 0.f);
    float ph = fmaxf(a3, 0.f);
    float cp = cosf(ph), sp = sinf(ph);

    float nt = sqrtf(blockReduceSum128(a0 * a0, sbuf));
    float na = sqrtf(blockReduceSum128(a1 * a1, sbuf));
    float nv = sqrtf(blockReduceSum128(a2 * a2, sbuf));
    float at = a0 / fmaxf(nt, EPSF);
    float aa = a1 / fmaxf(na, EPSF);
    float av = a2 / fmaxf(nv, EPSF);

    float mx = fmaxf(nt, fmaxf(na, nv));
    float e0 = __expf(nt - mx), e1 = __expf(na - mx), e2 = __expf(nv - mx);
    float es = e0 + e1 + e2;

    float q0r = at * cp, q0i = at * sp;
    float q1r = aa * cp, q1i = aa * sp;
    float q2r = av * cp, q2i = av * sp;
    float la1d = la[d], la2d = la[DD + d];
    float s1r = la1d * cp, s1i = la1d * sp;
    float s2r = la2d * cp, s2i = la2d * sp;
    float hr = (s1r + s1i) * INV_SQRT2F, hi = (s1r - s1i) * INV_SQRT2F;
    float p4r = hr * s2r - hi * s2i, p4i = hr * s2i + hi * s2r;
    float n4 = sqrtf(blockReduceSum128(p4r * p4r + p4i * p4i, sbuf));
    float q3r = p4r / fmaxf(n4, EPSF), q3i = p4i / fmaxf(n4, EPSF);

    float pr[3], pi[3];
    pr[0] = q0r * q1r - q0i * q1i;  pi[0] = q0r * q1i + q0i * q1r;
    pr[1] = q1r * q2r - q1i * q2i;  pi[1] = q1r * q2i + q1i * q2r;
    pr[2] = q2r * q3r - q2i * q3i;  pi[2] = q2r * q3i + q2i * q3r;
    #pragma unroll
    for (int m = 0; m < 3; ++m) {
        float nm = sqrtf(blockReduceSum128(pr[m] * pr[m] + pi[m] * pi[m], sbuf));
        float inv = 1.f / fmaxf(nm, EPSF);
        entr[(size_t)m * NBT * DD + (size_t)bt * DD + d] = pr[m] * inv;
        enti[(size_t)m * NBT * DD + (size_t)bt * DD + d] = pi[m] * inv;
    }
    if (d == 0) {
        wmod[bt * 3 + 0] = e0 / es;
        wmod[bt * 3 + 1] = e1 / es;
        wmod[bt * 3 + 2] = e2 / es;
    }
}

// ---------------- conv weights: f32 [O][I][tap] -> bf16 [O][tap*128+i] ----------------
__global__ __launch_bounds__(256) void convert_weights(
    const float* __restrict__ w1, const float* __restrict__ w2,
    unsigned short* __restrict__ Wb) {
    int i = blockIdx.x * blockDim.x + threadIdx.x;
    if (i >= 2 * DD * CK) return;
    int layer = i / (DD * CK);
    int r = i % (DD * CK);
    int o = r / CK, c = r % CK;
    int tap = c >> 7, ii = c & 127;
    const float* w = layer ? w2 : w1;
    Wb[i] = f2bf(w[o * CK + ii * 3 + tap]);
}

// ---------------- normalize + convert measurement vectors ----------------
__global__ __launch_bounds__(128) void convert_meas(
    const float* __restrict__ mr, const float* __restrict__ mi,
    unsigned short* __restrict__ Vb) {
    __shared__ float sbuf[128];
    int k = blockIdx.x, e = threadIdx.x;
    float r = mr[k * DD + e], im = mi[k * DD + e];
    float n = sqrtf(blockReduceSum128(r * r + im * im, sbuf));
    n = fmaxf(n, EPSF);
    Vb[k * DD + e]        = f2bf(r / n);
    Vb[SAMP + k * DD + e] = f2bf(im / n);
}

// ---------------- conv1d as implicit GEMM via WMMA bf16 ----------------
// xsl[tap][i][l] = x[i][l + tap - 1]  (zero at edges); B-frags 16B-aligned.
__global__ __launch_bounds__(256) void conv_wmma_kernel(
    const unsigned short* __restrict__ Wb,   // [128][384] bf16, c = tap*128+i
    const float* __restrict__ bias,
    unsigned short* __restrict__ act,        // [1600][128][128] bf16 in/out
    const float* __restrict__ entr, const float* __restrict__ enti,
    const float* __restrict__ wmod,
    int build_rho) {
    __shared__ unsigned short xsl[3][DD][DD];
    __shared__ float evr[3][DD], evi[3][DD], shw[3];

    int tid = threadIdx.x;
    int s = blockIdx.x;

    if (build_rho) {
        int bt = s % NBT, part = s / NBT;   // 0 = real, 1 = imag
        for (int i = tid; i < 3 * DD; i += 256) {
            int m = i >> 7, d = i & 127;
            evr[m][d] = entr[(size_t)m * NBT * DD + (size_t)bt * DD + d];
            evi[m][d] = enti[(size_t)m * NBT * DD + (size_t)bt * DD + d];
        }
        if (tid < 3) shw[tid] = wmod[bt * 3 + tid];
        __syncthreads();
        // center copy (tap=1 == unshifted sample) from rank-6 factors
        for (int idx = tid; idx < SAMP; idx += 256) {
            int i = idx >> 7, l = idx & 127;
            float val = 0.f;
            if (part == 0) {
                #pragma unroll
                for (int m = 0; m < 3; ++m)
                    val += shw[m] * (evr[m][i] * evr[m][l] + evi[m][i] * evi[m][l]);
            } else {
                #pragma unroll
                for (int m = 0; m < 3; ++m)
                    val += shw[m] * (evi[m][i] * evr[m][l] - evr[m][i] * evi[m][l]);
            }
            xsl[1][i][l] = f2bf(val);
        }
    } else {
        // async-stage the sample into the center copy
        const unsigned short* src = act + (size_t)s * SAMP;
        for (int c = tid; c < SAMP / 8; c += 256)
            async_b128(&xsl[1][0][0] + c * 8, src + c * 8);
        async_wait0();
    }
    __syncthreads();
    // shifted copies from LDS center copy
    for (int idx = tid; idx < SAMP; idx += 256) {
        int i = idx >> 7, l = idx & 127;
        xsl[0][i][l] = (l == 0)   ? (unsigned short)0 : xsl[1][i][l - 1];
        xsl[2][i][l] = (l == 127) ? (unsigned short)0 : xsl[1][i][l + 1];
    }
    __syncthreads();

    int wv = tid >> 5, lane = tid & 31;
    int half = lane >> 4, lm = lane & 15;
    int m0 = wv * 16;

    const unsigned short* wrow = Wb + (size_t)(m0 + lm) * CK;

    v8f zero = {};
    v8f acc[8];
    #pragma unroll
    for (int n = 0; n < 8; ++n) acc[n] = zero;

    for (int kc = 0; kc < CK / 32; ++kc) {
        BF16x16 af;
        af.q[0] = *(const v4u*)(wrow + kc * 32 + half * 8);        // global_load_b128
        af.q[1] = *(const v4u*)(wrow + kc * 32 + 16 + half * 8);
        int kg = kc * 32;
        int tap = kg >> 7, bi = kg & 127;
        const unsigned short* xrow = &xsl[tap][bi][0];
        #pragma unroll
        for (int nt = 0; nt < 8; ++nt) {
            BF16x16 bf;
            #pragma unroll
            for (int j = 0; j < 16; ++j) {
                int kk = (j < 8) ? (half * 8 + j) : (16 + half * 8 + (j - 8));
                bf.u[j] = xrow[kk * DD + nt * 16 + lm];
            }
            acc[nt] = __builtin_amdgcn_wmma_f32_16x16x32_bf16(
                false, af.v, false, bf.v, (short)0, acc[nt], false, false);
        }
    }

    unsigned short* dst = act + (size_t)s * SAMP;
    float bloc[8];
    #pragma unroll
    for (int g = 0; g < 8; ++g) bloc[g] = bias[m0 + half * 8 + g];
    #pragma unroll
    for (int nt = 0; nt < 8; ++nt) {
        #pragma unroll
        for (int g = 0; g < 8; ++g) {
            int row = m0 + half * 8 + g;
            int col = nt * 16 + lm;
            float v = fmaxf(acc[nt][g] + bloc[g], 0.f);
            dst[row * DD + col] = f2bf(v);
        }
    }
}

// ---------------- QMeasurement via WMMA ----------------
__global__ __launch_bounds__(256) void meas_wmma_kernel(
    const unsigned short* __restrict__ act,
    const unsigned short* __restrict__ Vb,
    float* __restrict__ pout) {
    __shared__ unsigned short xsm[2][DD][DD];
    __shared__ unsigned short vl[2][DD][DD];
    __shared__ float pl[DD];

    int tid = threadIdx.x, bt = blockIdx.x;
    const unsigned short* srcR = act + (size_t)bt * SAMP;
    const unsigned short* srcI = act + (size_t)(NBT + bt) * SAMP;
    unsigned short* x0 = &xsm[0][0][0];
    unsigned short* x1 = &xsm[1][0][0];
    unsigned short* v0 = &vl[0][0][0];
    unsigned short* v1 = &vl[1][0][0];
    for (int c = tid; c < SAMP / 8; c += 256) {
        async_b128(x0 + c * 8, srcR + c * 8);
        async_b128(x1 + c * 8, srcI + c * 8);
        async_b128(v0 + c * 8, Vb + c * 8);
        async_b128(v1 + c * 8, Vb + SAMP + c * 8);
    }
    if (tid < DD) pl[tid] = 0.f;
    async_wait0();
    __syncthreads();

    int wv = tid >> 5, lane = tid & 31;
    int half = lane >> 4, lm = lane & 15;

    for (int job = wv; job < 32; job += 8) {
        int src = job >> 4;        // 0 -> rho_r, 1 -> rho_i
        int mt = job & 15;         // tile over stacked [Vr; Vi] rows (M=256)
        int m0 = mt * 16;
        v8f zero = {};
        v8f acc[8];
        #pragma unroll
        for (int n = 0; n < 8; ++n) acc[n] = zero;

        int arow = m0 + lm;
        const unsigned short* vbase = (arow < DD) ? &vl[0][arow][0] : &vl[1][arow - DD][0];
        for (int kc = 0; kc < 4; ++kc) {
            BF16x16 af;
            af.q[0] = *(const v4u*)(vbase + kc * 32 + half * 8);
            af.q[1] = *(const v4u*)(vbase + kc * 32 + 16 + half * 8);
            int kg = kc * 32;
            const unsigned short* xb = (src == 0) ? &xsm[0][kg][0] : &xsm[1][kg][0];
            #pragma unroll
            for (int nt = 0; nt < 8; ++nt) {
                BF16x16 bf;
                #pragma unroll
                for (int j = 0; j < 16; ++j) {
                    int kk = (j < 8) ? (half * 8 + j) : (16 + half * 8 + (j - 8));
                    bf.u[j] = xb[kk * DD + nt * 16 + lm];
                }
                acc[nt] = __builtin_amdgcn_wmma_f32_16x16x32_bf16(
                    false, af.v, false, bf.v, (short)0, acc[nt], false, false);
            }
        }
        // p[m%128] += sum_n C[m,n]*sel[m,n]; cross-lane reduce over the 16 lanes sharing a row
        #pragma unroll
        for (int g = 0; g < 8; ++g) {
            int m = m0 + half * 8 + g;
            float partial = 0.f;
            #pragma unroll
            for (int nt = 0; nt < 8; ++nt) {
                int col = nt * 16 + lm;
                float sel;
                if (src == 0) sel = (m < DD) ?  bf2f(vl[0][m][col]) : bf2f(vl[1][m - DD][col]);
                else          sel = (m < DD) ? -bf2f(vl[1][m][col]) : bf2f(vl[0][m - DD][col]);
                partial += acc[nt][g] * sel;
            }
            #pragma unroll
            for (int ofs = 1; ofs < 16; ofs <<= 1)
                partial += __shfl_xor(partial, ofs, 32);
            if (lm == 0) atomicAdd(&pl[m & (DD - 1)], partial);
        }
    }
    __syncthreads();
    if (tid < DD) pout[(size_t)bt * DD + tid] = pl[tid];
}

// ---------------- head ----------------
__global__ __launch_bounds__(64) void head_kernel(
    const float* __restrict__ p, const float* __restrict__ w1, const float* __restrict__ b1,
    const float* __restrict__ w2, const float* __restrict__ b2, float* __restrict__ out) {
    __shared__ float hs[64];
    __shared__ float os[2];
    int bt = blockIdx.x, c = threadIdx.x;
    const float* pr = p + (size_t)bt * DD;
    float a = b1[c];
    for (int k = 0; k < DD; ++k) a = fmaf(pr[k], w1[k * 64 + c], a);
    hs[c] = fmaxf(a, 0.f);
    __syncthreads();
    if (c < 2) {
        float o = b2[c];
        for (int k = 0; k < 64; ++k) o = fmaf(hs[k], w2[k * 2 + c], o);
        os[c] = tanhf(o);
    }
    __syncthreads();
    if (c == 0) {
        float m = fmaxf(os[0], os[1]);
        float l = m + logf(__expf(os[0] - m) + __expf(os[1] - m));
        out[bt * 2 + 0] = os[0] - l;
        out[bt * 2 + 1] = os[1] - l;
    }
}

extern "C" void kernel_launch(void* const* d_in, const int* in_sizes, int n_in,
                              void* d_out, int out_size, void* d_ws, size_t ws_size,
                              hipStream_t stream) {
    (void)in_sizes; (void)n_in; (void)out_size; (void)ws_size;

    const float* text    = (const float*)d_in[0];
    const float* audio   = (const float*)d_in[1];
    const float* visual  = (const float*)d_in[2];
    const float* beats   = (const float*)d_in[3];
    const float* W_text  = (const float*)d_in[4];
    const float* b_text  = (const float*)d_in[5];
    const float* W_audio = (const float*)d_in[6];
    const float* b_audio = (const float*)d_in[7];
    const float* W_vis   = (const float*)d_in[8];
    const float* b_vis   = (const float*)d_in[9];
    const float* W_beats = (const float*)d_in[10];
    const float* b_beats = (const float*)d_in[11];
    const float* lang1   = (const float*)d_in[12];
    const float* lang2   = (const float*)d_in[13];
    const float* W_l1    = (const float*)d_in[14];
    const float* b_l1    = (const float*)d_in[15];
    const float* W_l2    = (const float*)d_in[16];
    const float* b_l2    = (const float*)d_in[17];
    const float* conv_w1 = (const float*)d_in[18];
    const float* conv_b1 = (const float*)d_in[19];
    const float* conv_w2 = (const float*)d_in[20];
    const float* conv_b2 = (const float*)d_in[21];
    const float* meas_r  = (const float*)d_in[22];
    const float* meas_i  = (const float*)d_in[23];
    const float* fc_w1   = (const float*)d_in[24];
    const float* fc_b1   = (const float*)d_in[25];
    const float* fc_w2   = (const float*)d_in[26];
    const float* fc_b2   = (const float*)d_in[27];
    float* out = (float*)d_out;

    char* base = (char*)d_ws;
    size_t off = 0;
    auto carve = [&](size_t bytes) {
        void* ptr = base + off;
        off += (bytes + 255) & ~(size_t)255;
        return ptr;
    };
    unsigned short* act  = (unsigned short*)carve((size_t)1600 * SAMP * 2);
    float* entr          = (float*)carve((size_t)3 * NBT * DD * 4);
    float* enti          = (float*)carve((size_t)3 * NBT * DD * 4);
    float* wmod          = (float*)carve((size_t)NBT * 3 * 4);
    float* la            = (float*)carve((size_t)2 * DD * 4);
    unsigned short* Wb   = (unsigned short*)carve((size_t)2 * DD * CK * 2);
    unsigned short* Vb   = (unsigned short*)carve((size_t)2 * SAMP * 2);
    float* pbuf          = (float*)carve((size_t)NBT * DD * 4);

    lang_kernel<<<1, 128, 0, stream>>>(lang1, lang2, W_l1, b_l1, W_l2, b_l2, la);
    prep_kernel<<<NBT, 128, 0, stream>>>(text, audio, visual, beats,
                                         W_text, b_text, W_audio, b_audio,
                                         W_vis, b_vis, W_beats, b_beats,
                                         la, entr, enti, wmod);
    convert_weights<<<(2 * DD * CK + 255) / 256, 256, 0, stream>>>(conv_w1, conv_w2, Wb);
    convert_meas<<<DD, 128, 0, stream>>>(meas_r, meas_i, Vb);

    conv_wmma_kernel<<<1600, 256, 0, stream>>>(Wb, conv_b1, act, entr, enti, wmod, 1);
    conv_wmma_kernel<<<1600, 256, 0, stream>>>(Wb + DD * CK, conv_b2, act, entr, enti, wmod, 0);

    meas_wmma_kernel<<<NBT, 256, 0, stream>>>(act, Vb, pbuf);
    head_kernel<<<NBT, 64, 0, stream>>>(pbuf, fc_w1, fc_b1, fc_w2, fc_b2, out);
}
